// NeuralODEDecoder_77936476553390
// MI455X (gfx1250) — matrix operand order repaired and checked by reading, not compiled, not run on hardware
//
#include <hip/hip_runtime.h>
#include <hip/hip_bf16.h>
#include <math.h>

typedef _Float16 v16h __attribute__((ext_vector_type(16)));
typedef _Float16 v8h  __attribute__((ext_vector_type(8)));
typedef float    v8f  __attribute__((ext_vector_type(8)));

#define S_STEPS 8
#define BATCH   8
#define DIM     11
#define H       128
#define NGRID   4096
#define ROWS    (S_STEPS*BATCH*NGRID)   // 262144
#define NTILES  (ROWS/16)               // 16384
#define NSUB    100
#define ZS_N    (S_STEPS*BATCH*DIM)     // 704

// Branch-free tanh: prefer gfx1250 hardware V_TANH_F32; fallback is a
// clamped exp-based rational (no EXEC divergence, unlike libm tanhf).
__device__ __forceinline__ float fast_tanh(float x) {
#if __has_builtin(__builtin_amdgcn_tanhf)
    return __builtin_amdgcn_tanhf(x);
#elif __has_builtin(__builtin_amdgcn_tanh_f32)
    return __builtin_amdgcn_tanh_f32(x);
#else
    x = fminf(fmaxf(x, -9.f), 9.f);
    const float t = __expf(2.f * x);
    return __fdividef(t - 1.f, t + 1.f);
#endif
}

// Wave-local LDS ordering: same-wave DS ops complete in order (DScnt), so a
// dscnt-0 wait makes prior LDS stores visible to subsequent same-wave loads.
__device__ __forceinline__ void wave_lds_fence() {
    asm volatile("s_wait_dscnt 0" ::: "memory");
}

// ---------------------------------------------------------------------------
// Kernel 1: RK4 neural-ODE scan. One workgroup, 1024 threads (32 wave32s).
// Thread (b = tid>>7, j = tid&127) owns hidden unit j of batch row b.
// All weights + state live in LDS (~86 KB of the 320 KB WGP LDS).
// ---------------------------------------------------------------------------
#define O_W1   0          // 12*128
#define O_B1   1536
#define O_W2   1664       // 128*128
#define O_B2   18048
#define O_W3   18176      // 128*11
#define O_B3   19584      // 11 (padded)
#define O_Y    19600      // 88 (padded to 96)
#define O_YT   19696
#define O_KA   19792
#define O_KC   19888
#define O_H1   19984      // 8*128
#define O_H2   21008      // 8*128
#define ODE_LDS_FLOATS 22032

__device__ __forceinline__ void odef(float* lds, int stateOff, float tt, int outOff, int tid) {
    const int b = tid >> 7;
    const int j = tid & (H - 1);
    // layer 1: (8x12) @ (12x128), xt = [y, t]
    float acc = lds[O_B1 + j];
    const float* st = lds + stateOff + b * DIM;
    #pragma unroll
    for (int k = 0; k < DIM; ++k) acc += st[k] * lds[O_W1 + k * H + j];
    acc += tt * lds[O_W1 + DIM * H + j];
    lds[O_H1 + b * H + j] = fmaxf(acc, 0.f);
    __syncthreads();
    // layer 2: (8x128) @ (128x128)
    acc = lds[O_B2 + j];
    #pragma unroll 8
    for (int k = 0; k < H; ++k) acc += lds[O_H1 + b * H + k] * lds[O_W2 + k * H + j];
    lds[O_H2 + b * H + j] = fmaxf(acc, 0.f);
    __syncthreads();
    // layer 3: (8x128) @ (128x11) -> 88 outputs, one per thread
    if (tid < BATCH * DIM) {
        const int bb = tid / DIM, d = tid - bb * DIM;
        float a2 = lds[O_B3 + d];
        #pragma unroll 8
        for (int k = 0; k < H; ++k) a2 += lds[O_H2 + bb * H + k] * lds[O_W3 + k * DIM + d];
        lds[outOff + tid] = a2;
    }
    __syncthreads();
}

__global__ __launch_bounds__(1024) void ode_kernel(
    const float* __restrict__ z0, const float* __restrict__ tarr,
    const float* __restrict__ w1, const float* __restrict__ b1,
    const float* __restrict__ w2, const float* __restrict__ b2,
    const float* __restrict__ w3, const float* __restrict__ b3,
    float* __restrict__ zs_out) {
    extern __shared__ float lds[];
    const int tid = threadIdx.x;
    for (int i = tid; i < 12 * H;  i += 1024) lds[O_W1 + i] = w1[i];
    for (int i = tid; i < H * H;   i += 1024) lds[O_W2 + i] = w2[i];
    for (int i = tid; i < H * DIM; i += 1024) lds[O_W3 + i] = w3[i];
    if (tid < H) { lds[O_B1 + tid] = b1[tid]; lds[O_B2 + tid] = b2[tid]; }
    if (tid < DIM) lds[O_B3 + tid] = b3[tid];
    if (tid < BATCH * DIM) { float v = z0[tid]; lds[O_Y + tid] = v; zs_out[tid] = v; }
    __syncthreads();

    for (int iv = 0; iv < S_STEPS - 1; ++iv) {
        const float t0 = tarr[iv], t1 = tarr[iv + 1];
        const float dt = (t1 - t0) * (1.f / NSUB);
        for (int s = 0; s < NSUB; ++s) {
            const float tt = t0 + (float)s * dt;
            odef(lds, O_Y, tt, O_KC, tid);
            if (tid < BATCH * DIM) {
                float k1 = lds[O_KC + tid];
                lds[O_KA + tid] = k1;
                lds[O_YT + tid] = lds[O_Y + tid] + 0.5f * dt * k1;
            }
            __syncthreads();
            odef(lds, O_YT, tt + 0.5f * dt, O_KC, tid);
            if (tid < BATCH * DIM) {
                float k2 = lds[O_KC + tid];
                lds[O_KA + tid] += 2.f * k2;
                lds[O_YT + tid] = lds[O_Y + tid] + 0.5f * dt * k2;
            }
            __syncthreads();
            odef(lds, O_YT, tt + 0.5f * dt, O_KC, tid);
            if (tid < BATCH * DIM) {
                float k3 = lds[O_KC + tid];
                lds[O_KA + tid] += 2.f * k3;
                lds[O_YT + tid] = lds[O_Y + tid] + dt * k3;
            }
            __syncthreads();
            odef(lds, O_YT, tt + dt, O_KC, tid);
            if (tid < BATCH * DIM)
                lds[O_Y + tid] += (dt * (1.f / 6.f)) * (lds[O_KA + tid] + lds[O_KC + tid]);
            __syncthreads();
        }
        if (tid < BATCH * DIM) zs_out[(iv + 1) * BATCH * DIM + tid] = lds[O_Y + tid];
        __syncthreads();
    }
}

// ---------------------------------------------------------------------------
// Kernel 2: analytic BatchNorm statistics + fold into first-layer affine.
// h = c0*w0 + c1*w1 + b is affine in coords; grid moments are closed-form,
// so per-channel mean/var over all 262144 rows come from the 64 (s,b) states.
// ws layout: [0..255] rot params (cs,sn,dx0,dx1)*64 ; [256..639] eff w0/w1/b.
// ---------------------------------------------------------------------------
__global__ __launch_bounds__(128) void stats_kernel(
    const float* __restrict__ zs, const float* __restrict__ dx_prior,
    const float* __restrict__ cw, const float* __restrict__ cb,
    const float* __restrict__ gamma, const float* __restrict__ beta,
    float* __restrict__ ws) {
    __shared__ float mom[5][64];
    __shared__ float red[5];
    const int tid = threadIdx.x;
    if (tid < 64) {
        const float theta = zs[tid * DIM + 0];
        const float dxp = dx_prior[0];
        const float dx0 = zs[tid * DIM + 1] * dxp;
        const float dx1 = zs[tid * DIM + 2] * dxp;
        const float cs = cosf(theta), sn = sinf(theta);
        ws[tid * 4 + 0] = cs; ws[tid * 4 + 1] = sn;
        ws[tid * 4 + 2] = dx0; ws[tid * 4 + 3] = dx1;
        mom[0][tid] = dx0;        // E[c0]
        mom[1][tid] = dx1;        // E[c1]
        mom[2][tid] = dx0 * dx0;  // E[c0^2] - c2
        mom[3][tid] = dx1 * dx1;  // E[c1^2] - c2
        mom[4][tid] = dx0 * dx1;  // E[c0*c1]
    }
    __syncthreads();
    if (tid < 5) {
        float s = 0.f;
        for (int i = 0; i < 64; ++i) s += mom[tid][i];
        red[tid] = s * (1.f / 64.f);
    }
    __syncthreads();
    float c2 = 0.f;   // mean of linspace(-1,1,64)^2
    for (int i = 0; i < 64; ++i) {
        float x = -1.f + (2.f / 63.f) * (float)i;
        c2 += x * x;
    }
    c2 *= (1.f / 64.f);
    const float M1x = red[0], M1y = red[1];
    const float M2x = red[2] + c2, M2y = red[3] + c2, Mxy = red[4];
    const float w0 = cw[tid], w1 = cw[H + tid], b = cb[tid];
    const float mu = w0 * M1x + w1 * M1y + b;
    const float e2 = w0 * w0 * M2x + w1 * w1 * M2y + 2.f * w0 * w1 * Mxy
                   + 2.f * b * (w0 * M1x + w1 * M1y) + b * b;
    const float var = e2 - mu * mu;
    const float g = gamma[tid] * rsqrtf(var + 1e-5f);
    float* eff = ws + 256;
    eff[tid]         = w0 * g;
    eff[H + tid]     = w1 * g;
    eff[2 * H + tid] = (b - mu) * g + beta[tid];
}

// ---------------------------------------------------------------------------
// Kernel 3: decoder MLP via v_wmma_f32_16x16x32_f16.
// 4 waves/block, each wave owns a 16-row tile at a time. Weights (3x 128x128,
// f16, transposed) live in LDS; activations bounce through a PER-WAVE LDS
// tile between layers, so only wave-local dscnt fences are needed (no
// cross-wave __syncthreads in the hot loop).
// ---------------------------------------------------------------------------
#define D_WT   0                         // 3*128*128 f16 = 98304 B
#define D_OW   98304                     // 128 f32
#define D_LB   98816                     // 3*128 f32
#define D_EFF  100352                    // 3*128 f32
#define D_ACT  101888                    // 4 waves * 16*128 f16 = 16384 B
#define DEC_LDS_BYTES 118272

__global__ __launch_bounds__(128) void decoder_kernel(
    const float* __restrict__ ws,
    const float* __restrict__ l0w, const float* __restrict__ l0b,
    const float* __restrict__ l1w, const float* __restrict__ l1b,
    const float* __restrict__ l2w, const float* __restrict__ l2b,
    const float* __restrict__ outw, const float* __restrict__ outbp,
    float* __restrict__ out) {
    extern __shared__ char smem[];
    _Float16* wT      = (_Float16*)(smem + D_WT);
    float*    owf     = (float*)(smem + D_OW);
    float*    lb      = (float*)(smem + D_LB);
    float*    eff     = (float*)(smem + D_EFF);
    _Float16* actbase = (_Float16*)(smem + D_ACT);
    const int tid = threadIdx.x;

    // Hint the weight streams into cache (global_prefetch_b8 on gfx1250).
    __builtin_prefetch(l0w + tid * 64, 0, 1);
    __builtin_prefetch(l1w + tid * 64, 0, 1);
    __builtin_prefetch(l2w + tid * 64, 0, 1);

    // Load weights transposed (wT[layer][n][k]) so a B-fragment is one
    // contiguous 32B v16h per lane.
    for (int idx = tid; idx < 3 * H * H; idx += 128) {
        const int l = idx / (H * H);
        const int e = idx - l * (H * H);
        const int k = e >> 7, n = e & (H - 1);
        const float* w = (l == 0) ? l0w : ((l == 1) ? l1w : l2w);
        wT[l * H * H + n * H + k] = (_Float16)w[e];
    }
    lb[tid] = l0b[tid]; lb[H + tid] = l1b[tid]; lb[2 * H + tid] = l2b[tid];
    owf[tid] = outw[tid];
    eff[tid] = ws[256 + tid];
    eff[H + tid] = ws[256 + H + tid];
    eff[2 * H + tid] = ws[256 + 2 * H + tid];
    __syncthreads();

    const int wave = tid >> 5, lane = tid & 31;
    const int m = lane & 15, half = lane >> 4;
    _Float16* act = actbase + wave * (16 * H);
    const float outb = outbp[0];
    const int nwaves = gridDim.x * 4;   // 16384 tiles / 4096 waves = 4 each

    for (int tile = blockIdx.x * 4 + wave; tile < NTILES; tile += nwaves) {
        const int row0 = tile * 16;
        const int sb = row0 >> 12;                // 4096 rows per (s,b)
        const float cs = ws[sb * 4 + 0], sn = ws[sb * 4 + 1];
        const float dx0 = ws[sb * 4 + 2], dx1 = ws[sb * 4 + 3];
        const int n = (row0 + m) & (NGRID - 1);
        const float gx = -1.f + (2.f / 63.f) * (float)(n >> 6);
        const float gy = -1.f + (2.f / 63.f) * (float)(n & 63);
        const float c0 =  gx * cs + gy * sn + dx0;
        const float c1 = -gx * sn + gy * cs + dx1;

        // Layer 0 (2 -> 128) + folded BatchNorm, straight VALU.
        #pragma unroll 4
        for (int j = half * 64; j < half * 64 + 64; ++j) {
            const float h = c0 * eff[j] + c1 * eff[H + j] + eff[2 * H + j];
            act[m * H + j] = (_Float16)h;
        }
        wave_lds_fence();

        // Layers 1..3: 16x128 @ 128x128 + tanh, via WMMA (32 wmma each).
        #pragma unroll
        for (int layer = 0; layer < 3; ++layer) {
            // A fragments: contiguous k order, matching the B fragments
            //  lanes 0-15 (M=lane): K {0..15}; lanes 16-31: K {16..31}
            v16h afr[4];
            const int koffb = half ? 8 : 0;
            #pragma unroll
            for (int kc = 0; kc < 4; ++kc) {
                afr[kc] = *(const v16h*)(act + m * H + kc * 32 + half * 16);
            }
            const _Float16* wl = wT + layer * H * H;
            #pragma unroll
            for (int nt = 0; nt < 8; ++nt) {
                const int col = nt * 16 + m;
                const float bj = lb[layer * H + col];
                v8f c = {bj, bj, bj, bj, bj, bj, bj, bj};
                #pragma unroll
                for (int kc = 0; kc < 4; ++kc) {
                    // B fragment: lanes 0-15 -> K kc*32..+15 of column `col`,
                    // lanes 16-31 -> K kc*32+16..+31 (contiguous in wT).
                    v16h bfr = *(const v16h*)(wl + col * H + kc * 32 + half * 16);
                    c = __builtin_amdgcn_wmma_f32_16x16x32_f16(
                            false, afr[kc], false, bfr, (short)0, c, false, false);
                }
                // C layout: lane -> N=col, VGPR r -> M=r (+8 for upper half)
                #pragma unroll
                for (int r = 0; r < 8; ++r) {
                    const int mr = half ? (r + 8) : r;
                    act[mr * H + col] = (_Float16)fast_tanh(c[r]);
                }
            }
            wave_lds_fence();
        }

        // Output layer 128 -> 1 (per-row dot product, cross-half shuffle).
        float s = 0.f;
        #pragma unroll 8
        for (int j = 0; j < 64; ++j)
            s += (float)act[m * H + half * 64 + j] * owf[half * 64 + j];
        s += __shfl_xor(s, 16, 32);
        if (half == 0) out[row0 + m] = s + outb;
        wave_lds_fence();
    }
}

// ---------------------------------------------------------------------------
extern "C" void kernel_launch(void* const* d_in, const int* in_sizes, int n_in,
                              void* d_out, int out_size, void* d_ws, size_t ws_size,
                              hipStream_t stream) {
    const float* z0    = (const float*)d_in[0];
    const float* t     = (const float*)d_in[1];
    const float* dxp   = (const float*)d_in[2];
    const float* ow1   = (const float*)d_in[3];
    const float* ob1   = (const float*)d_in[4];
    const float* ow2   = (const float*)d_in[5];
    const float* ob2   = (const float*)d_in[6];
    const float* ow3   = (const float*)d_in[7];
    const float* ob3   = (const float*)d_in[8];
    const float* cw    = (const float*)d_in[9];
    const float* cb    = (const float*)d_in[10];
    const float* gamma = (const float*)d_in[11];
    const float* beta  = (const float*)d_in[12];
    const float* l0w   = (const float*)d_in[13];
    const float* l0b   = (const float*)d_in[14];
    const float* l1w   = (const float*)d_in[15];
    const float* l1b   = (const float*)d_in[16];
    const float* l2w   = (const float*)d_in[17];
    const float* l2b   = (const float*)d_in[18];
    const float* outw  = (const float*)d_in[19];
    const float* outb  = (const float*)d_in[20];

    float* out    = (float*)d_out;
    float* zs_out = out + ROWS;          // tuple output: xs (262144) then zs (704)
    float* ws     = (float*)d_ws;

    ode_kernel<<<1, 1024, ODE_LDS_FLOATS * sizeof(float), stream>>>(
        z0, t, ow1, ob1, ow2, ob2, ow3, ob3, zs_out);
    stats_kernel<<<1, 128, 0, stream>>>(zs_out, dxp, cw, cb, gamma, beta, ws);
    decoder_kernel<<<1024, 128, DEC_LDS_BYTES, stream>>>(
        ws, l0w, l0b, l1w, l1b, l2w, l2b, outw, outb, out);
}
